// TacotronMelDecoder_5368709120362
// MI455X (gfx1250) — matrix operand order repaired and checked by reading, not compiled, not run on hardware
//
#include <hip/hip_runtime.h>

#define B_   32
#define T_   400
#define M_   200

typedef unsigned int  v4u   __attribute__((ext_vector_type(4)));
typedef float         v8f   __attribute__((ext_vector_type(8)));
typedef __bf16        v16bf __attribute__((ext_vector_type(16)));

struct BFPair { v4u lo, hi; };

static __device__ inline v16bf make_frag(v4u lo, v4u hi) {
  BFPair p{lo, hi};
  return __builtin_bit_cast(v16bf, p);
}

static __device__ inline unsigned short f2bf(float f) {
  unsigned int u = __float_as_uint(f);
  u += 0x7FFFu + ((u >> 16) & 1u);   // round-to-nearest-even
  return (unsigned short)(u >> 16);
}

static __device__ inline float sigm(float x) { return 1.0f / (1.0f + __expf(-x)); }

// A fragment: 16x32 bf16 from row-major [M][lda]; lane l holds row m0+(l&15),
// elements j<8 -> K = k0 + half*8 + j ; j>=8 -> K = k0 + 16 + half*8 + (j-8)
static __device__ inline v16bf load_a(const unsigned short* A, int lda, int m, int half, int k0) {
  const unsigned short* p = A + (size_t)m * lda + k0 + half * 8;
  return make_frag(*(const v4u*)p, *(const v4u*)(p + 16));
}
// B fragment: 32x16 bf16 from column-major weight Wt[N][ldb]; lane l holds
// column n, elements j<8 -> K = k0 + half*8 + j ; j>=8 -> K = k0 + 16 + half*8 + (j-8)
static __device__ inline v16bf load_b(const unsigned short* Bt, int ldb, int n, int half, int k0) {
  const unsigned short* p = Bt + (size_t)n * ldb + k0 + half * 8;
  return make_frag(*(const v4u*)p, *(const v4u*)(p + 16));
}

static __device__ inline v8f wmma_bf16(v16bf a, v16bf b, v8f c) {
  return __builtin_amdgcn_wmma_f32_16x16x32_bf16(false, a, false, b, (short)0, c, false, false);
}

// ---------------------------------------------------------------- prep kernels
// f32 [K][N] row-major -> bf16 Wt[N][ldo] (column-major of W), zero-padded K->Kpad
__global__ void k_transpose_w(const float* __restrict__ W, unsigned short* __restrict__ Wt,
                              int K, int N, int ldo, int koff, int Kpad) {
  int idx = blockIdx.x * blockDim.x + threadIdx.x;
  if (idx >= N * Kpad) return;
  int n = idx / Kpad, k = idx % Kpad;
  float v = (k < K) ? W[(size_t)k * N + n] : 0.f;
  Wt[(size_t)n * ldo + koff + k] = f2bf(v);
}

// f32 [rows][cols] -> bf16 [rows][ldo] zero-padded in last dim
__global__ void k_cvt_pad(const float* __restrict__ in, unsigned short* __restrict__ out,
                          int rows, int cols, int ldo) {
  int idx = blockIdx.x * blockDim.x + threadIdx.x;
  if (idx >= rows * ldo) return;
  int r = idx / ldo, c = idx % ldo;
  out[idx] = (c < cols) ? f2bf(in[(size_t)r * cols + c]) : (unsigned short)0;
}

// ---------------------------------------------------------------- generic GEMM
// C[M][N] = act(A[M][K] * Wt^T + bias); A bf16 row-major (lda=K), Wt bf16 [N][ldb]
// block = 256 thr (8 waves); wave -> 16 rows x 64 cols (A-frag reused over 4 N-tiles)
__global__ void __launch_bounds__(256) k_gemm_bf16(
    const unsigned short* __restrict__ A, int lda,
    const unsigned short* __restrict__ Bt, int ldb,
    const float* __restrict__ bias,
    float* __restrict__ Cf, unsigned short* __restrict__ Cb,
    int M, int N, int K, int act) {
  int lane = threadIdx.x & 31;
  int w    = threadIdx.x >> 5;
  int m0   = (blockIdx.y * 8 + w) * 16;
  if (m0 >= M) return;
  int half = lane >> 4;
  int nlo  = lane & 15;
  int ma   = m0 + nlo;
  int nb   = blockIdx.x * 64;
  int mb   = m0 + (half << 3);

  if (nb + 64 <= N) {
    // ---- fast path: 4 full N-tiles, branch-free inner loop (A-frag reused 4x)
    v8f acc[4];
#pragma unroll
    for (int i = 0; i < 4; ++i) acc[i] = (v8f){0.f,0.f,0.f,0.f,0.f,0.f,0.f,0.f};
    for (int k0 = 0; k0 < K; k0 += 32) {
      v16bf af = load_a(A, lda, ma, half, k0);
#pragma unroll
      for (int i = 0; i < 4; ++i) {
        v16bf bf = load_b(Bt, ldb, nb + i * 16 + nlo, half, k0);
        acc[i] = wmma_bf16(af, bf, acc[i]);
      }
    }
#pragma unroll
    for (int i = 0; i < 4; ++i) {
      int n = nb + i * 16 + nlo;
      float bv = bias ? bias[n] : 0.f;
#pragma unroll
      for (int r = 0; r < 8; ++r) {
        float v = acc[i][r] + bv;
        if (act) v = fmaxf(v, 0.f);
        size_t o = (size_t)(mb + r) * N + n;
        if (Cf) Cf[o] = v;
        if (Cb) Cb[o] = f2bf(v);
      }
    }
  } else {
    // ---- remainder path: tile-at-a-time (only the ragged N edge lands here)
    for (int i = 0; i < 4; ++i) {
      int n0 = nb + i * 16;
      if (n0 >= N) break;
      int n = n0 + nlo;
      v8f acc = {0.f,0.f,0.f,0.f,0.f,0.f,0.f,0.f};
      for (int k0 = 0; k0 < K; k0 += 32) {
        v16bf af = load_a(A, lda, ma, half, k0);
        v16bf bf = load_b(Bt, ldb, n < N ? n : (N - 1), half, k0);
        acc = wmma_bf16(af, bf, acc);
      }
      if (n < N) {
        float bv = bias ? bias[n] : 0.f;
#pragma unroll
        for (int r = 0; r < 8; ++r) {
          float v = acc[r] + bv;
          if (act) v = fmaxf(v, 0.f);
          size_t o = (size_t)(mb + r) * N + n;
          if (Cf) Cf[o] = v;
          if (Cb) Cb[o] = f2bf(v);
        }
      }
    }
  }
}

// -------------------------------------------------- attention + LSTM scan (1 WG)
// 32 waves; wave w owns batch b=w for the VALU stages; all waves cooperate on WMMA.
__global__ void __launch_bounds__(1024) k_lstm_attn_scan(
    const unsigned short* __restrict__ xp,      // [B,T,256] bf16
    const float* __restrict__ keys,             // [B,200,512] f32 (key_b applied)
    const float* __restrict__ attended,         // [B,200,256] f32
    const unsigned short* __restrict__ queryWt, // [512][512] bf16 (col-major)
    const float* __restrict__ query_b,          // [512]
    const float* __restrict__ attnvW,           // [512]
    const float* __restrict__ attnvb,           // [1]
    const unsigned short* __restrict__ lstmWt,  // [2048][1024] bf16 (K|R stacked)
    const float* __restrict__ lstm_b,           // [2048]
    float* __restrict__ zscr,                   // [32][2048] f32 scratch (L2)
    float* __restrict__ xF,                     // [B,T,512] f32 out
    unsigned short* __restrict__ xB)            // [B,T,512] bf16 out
{
  extern __shared__ char smem[];
  unsigned short* inpA = (unsigned short*)smem;                  // [32][1024] bf16: xt|ctx|h
  float* qbuf = (float*)(smem + 32 * 1024 * 2);                  // [32][512]
  float* sbuf = (float*)(smem + 32 * 1024 * 2 + 32 * 512 * 4);   // [32][200]

  int tid = threadIdx.x, lane = tid & 31, w = tid >> 5;
  int b = w;
  int half = lane >> 4, nlo = lane & 15;

  float creg[16];
#pragma unroll
  for (int j = 0; j < 16; ++j) creg[j] = 0.f;
#pragma unroll
  for (int j = 0; j < 16; ++j) inpA[b * 1024 + 512 + lane + 32 * j] = 0;  // h=0
  float vreg[16];
#pragma unroll
  for (int j = 0; j < 16; ++j) vreg[j] = attnvW[lane + 32 * j];
  float vb = attnvb[0];

#pragma unroll 1
  for (int t = 0; t < T_; ++t) {
    // ---- stage A: xt -> LDS cols [0,256)
    for (int j = lane; j < 256; j += 32)
      inpA[b * 1024 + j] = xp[((size_t)b * T_ + t) * 256 + j];
    __syncthreads();

    // ---- stage B: q = h @ query_W + b  (wave -> n-tile w; B-frag shared by 2 M-tiles)
    {
      int n = w * 16 + nlo;
      float bv = query_b[n];
      v8f a0 = {0.f,0.f,0.f,0.f,0.f,0.f,0.f,0.f};
      v8f a1 = {0.f,0.f,0.f,0.f,0.f,0.f,0.f,0.f};
      for (int k0 = 0; k0 < 512; k0 += 32) {
        v16bf bf = load_b(queryWt, 512, n, half, k0);
        v16bf f0 = load_a(inpA, 1024, nlo, half, 512 + k0);        // rows 0..15
        v16bf f1 = load_a(inpA, 1024, 16 + nlo, half, 512 + k0);   // rows 16..31
        a0 = wmma_bf16(f0, bf, a0);
        a1 = wmma_bf16(f1, bf, a1);
      }
      int mb = half << 3;
#pragma unroll
      for (int r = 0; r < 8; ++r) {
        qbuf[(size_t)(mb + r) * 512 + n]      = a0[r] + bv;
        qbuf[(size_t)(16 + mb + r) * 512 + n] = a1[r] + bv;
      }
    }
    __syncthreads();

    // ---- stage C: scores, softmax, context (wave w handles batch b)
    float qreg[16];
#pragma unroll
    for (int j = 0; j < 16; ++j) qreg[j] = qbuf[b * 512 + lane + 32 * j];
#pragma unroll 1
    for (int mm = 0; mm < M_; ++mm) {
      const float* kp = keys + ((size_t)b * M_ + mm) * 512;
      float s = 0.f;
#pragma unroll
      for (int j = 0; j < 16; ++j) {
        int u = lane + 32 * j;
        s += tanhf(kp[u] + qreg[j]) * vreg[j];
      }
      for (int off = 16; off > 0; off >>= 1) s += __shfl_xor(s, off, 32);
      if (lane == 0) sbuf[b * M_ + mm] = s + vb;
    }
    float mx = -1e30f;
    for (int i = lane; i < M_; i += 32) mx = fmaxf(mx, sbuf[b * M_ + i]);
    for (int off = 16; off > 0; off >>= 1) mx = fmaxf(mx, __shfl_xor(mx, off, 32));
    float ssum = 0.f;
    for (int i = lane; i < M_; i += 32) {
      float e = __expf(sbuf[b * M_ + i] - mx);
      sbuf[b * M_ + i] = e;
      ssum += e;
    }
    for (int off = 16; off > 0; off >>= 1) ssum += __shfl_xor(ssum, off, 32);
    float inv = 1.f / ssum;
#pragma unroll 1
    for (int jj = 0; jj < 8; ++jj) {
      int d = lane + 32 * jj;
      float a = 0.f;
#pragma unroll 1
      for (int mm = 0; mm < M_; ++mm)
        a += sbuf[b * M_ + mm] * attended[((size_t)b * M_ + mm) * 256 + d];
      inpA[b * 1024 + 256 + d] = f2bf(a * inv);        // ctx -> cols [256,512)
    }
    __syncthreads();

    // ---- stage D: z = [xt|ctx|h] @ [K;R] + b  (wave -> 4 N-tiles x 2 M-tiles)
#pragma unroll 1
    for (int i = 0; i < 4; ++i) {
      int n = (w + 32 * i) * 16 + nlo;
      float bv = lstm_b[n];
      v8f a0 = {0.f,0.f,0.f,0.f,0.f,0.f,0.f,0.f};
      v8f a1 = {0.f,0.f,0.f,0.f,0.f,0.f,0.f,0.f};
      for (int k0 = 0; k0 < 1024; k0 += 32) {
        v16bf bf = load_b(lstmWt, 1024, n, half, k0);
        v16bf f0 = load_a(inpA, 1024, nlo, half, k0);
        v16bf f1 = load_a(inpA, 1024, 16 + nlo, half, k0);
        a0 = wmma_bf16(f0, bf, a0);
        a1 = wmma_bf16(f1, bf, a1);
      }
      int mb = half << 3;
#pragma unroll
      for (int r = 0; r < 8; ++r) {
        zscr[(size_t)(mb + r) * 2048 + n]      = a0[r] + bv;
        zscr[(size_t)(16 + mb + r) * 2048 + n] = a1[r] + bv;
      }
    }
    __threadfence_block();
    __syncthreads();

    // ---- stage E: gates + state update (thread owns (b, u=lane+32j))
    {
      const float* zr = zscr + (size_t)b * 2048;
      size_t xo = ((size_t)b * T_ + t) * 512;
#pragma unroll
      for (int j = 0; j < 16; ++j) {
        int u = lane + 32 * j;
        float ig = sigm(zr[u]);
        float fg = sigm(zr[512 + u]);
        float gg = tanhf(zr[1024 + u]);
        float og = sigm(zr[1536 + u]);
        creg[j] = fg * creg[j] + ig * gg;
        float hn = og * tanhf(creg[j]);
        xF[xo + u] = hn;
        unsigned short hb = f2bf(hn);
        xB[xo + u] = hb;
        inpA[b * 1024 + 512 + u] = hb;
      }
    }
  }
}

// ------------------------------------------------------------- GRU scan (1 WG)
__global__ void __launch_bounds__(1024) k_gru_scan(
    const float* __restrict__ xz,          // [B,T,1536] (bi included)
    const unsigned short* __restrict__ Rt, // [1536][512] bf16 col-major
    const float* __restrict__ br,          // [1536]
    const float* __restrict__ baseF,       // [B,T,512] residual input
    float* __restrict__ recG,              // [32][1536] f32 scratch (L2)
    float* __restrict__ outF,              // optional [B,T,512] (base + y)
    unsigned short* __restrict__ outB)     // [B,T,512] bf16 (base + y)
{
  extern __shared__ char smem[];
  unsigned short* hb = (unsigned short*)smem;   // [32][512] bf16
  int tid = threadIdx.x, lane = tid & 31, w = tid >> 5;
  int b = w;
  int half = lane >> 4, nlo = lane & 15;
  float h[16];
#pragma unroll
  for (int j = 0; j < 16; ++j) h[j] = 0.f;

#pragma unroll 1
  for (int t = 0; t < T_; ++t) {
#pragma unroll
    for (int j = 0; j < 16; ++j) hb[b * 512 + lane + 32 * j] = f2bf(h[j]);
    __syncthreads();

    // rec = h @ R + br  (wave -> 3 N-tiles x 2 M-tiles; B-frag shared)
#pragma unroll 1
    for (int i = 0; i < 3; ++i) {
      int n = (w + 32 * i) * 16 + nlo;
      float bv = br[n];
      v8f a0 = {0.f,0.f,0.f,0.f,0.f,0.f,0.f,0.f};
      v8f a1 = {0.f,0.f,0.f,0.f,0.f,0.f,0.f,0.f};
      for (int k0 = 0; k0 < 512; k0 += 32) {
        v16bf bf = load_b(Rt, 512, n, half, k0);
        v16bf f0 = load_a(hb, 512, nlo, half, k0);
        v16bf f1 = load_a(hb, 512, 16 + nlo, half, k0);
        a0 = wmma_bf16(f0, bf, a0);
        a1 = wmma_bf16(f1, bf, a1);
      }
      int mb = half << 3;
#pragma unroll
      for (int r = 0; r < 8; ++r) {
        recG[(size_t)(mb + r) * 1536 + n]      = a0[r] + bv;
        recG[(size_t)(16 + mb + r) * 1536 + n] = a1[r] + bv;
      }
    }
    __threadfence_block();
    __syncthreads();

    const float* xzr = xz + ((size_t)b * T_ + t) * 1536;
    const float* rr  = recG + (size_t)b * 1536;
    size_t so = ((size_t)b * T_ + t) * 512;
#pragma unroll
    for (int j = 0; j < 16; ++j) {
      int u = lane + 32 * j;
      float zg = sigm(xzr[u] + rr[u]);
      float rg = sigm(xzr[512 + u] + rr[512 + u]);
      float hh = tanhf(xzr[1024 + u] + rg * rr[1024 + u]);
      h[j] = zg * h[j] + (1.f - zg) * hh;
      float s = baseF[so + u] + h[j];
      if (outF) outF[so + u] = s;
      outB[so + u] = f2bf(s);
    }
  }
}

// ------------------------------------------------------------------- launcher
extern "C" void kernel_launch(void* const* d_in, const int* in_sizes, int n_in,
                              void* d_out, int out_size, void* d_ws, size_t ws_size,
                              hipStream_t stream) {
  (void)in_sizes; (void)n_in; (void)out_size; (void)ws_size;
  const float* inputs   = (const float*)d_in[0];
  const float* attended = (const float*)d_in[1];
  const float* pre_W1   = (const float*)d_in[2];
  const float* pre_b1   = (const float*)d_in[3];
  const float* pre_W2   = (const float*)d_in[4];
  const float* pre_b2   = (const float*)d_in[5];
  const float* key_W    = (const float*)d_in[6];
  const float* key_b    = (const float*)d_in[7];
  const float* query_W  = (const float*)d_in[8];
  const float* query_b  = (const float*)d_in[9];
  const float* attnv_W  = (const float*)d_in[10];
  const float* attnv_b  = (const float*)d_in[11];
  const float* lstm_K   = (const float*)d_in[12];
  const float* lstm_R   = (const float*)d_in[13];
  const float* lstm_b   = (const float*)d_in[14];
  const float* gru1_K   = (const float*)d_in[15];
  const float* gru1_R   = (const float*)d_in[16];
  const float* gru1_b   = (const float*)d_in[17];
  const float* gru2_K   = (const float*)d_in[18];
  const float* gru2_R   = (const float*)d_in[19];
  const float* gru2_b   = (const float*)d_in[20];
  const float* proj_W   = (const float*)d_in[21];

  const int BT = B_ * T_;   // 12800
  const int BM = B_ * M_;   // 6400

  char* base = (char*)d_ws;
  size_t off = 0;
  auto alloc = [&](size_t bytes) -> void* {
    void* p = base + off;
    off += (bytes + 255) & ~(size_t)255;
    return p;
  };

  unsigned short* inp_pad = (unsigned short*)alloc((size_t)BT * 96 * 2);
  unsigned short* att_bf  = (unsigned short*)alloc((size_t)BM * 256 * 2);
  unsigned short* preW1t  = (unsigned short*)alloc((size_t)512 * 96 * 2);
  unsigned short* preW2t  = (unsigned short*)alloc((size_t)256 * 512 * 2);
  unsigned short* keyWt   = (unsigned short*)alloc((size_t)512 * 256 * 2);
  unsigned short* queryWt = (unsigned short*)alloc((size_t)512 * 512 * 2);
  unsigned short* lstmWt  = (unsigned short*)alloc((size_t)2048 * 1024 * 2);
  unsigned short* gru1Kt  = (unsigned short*)alloc((size_t)1536 * 512 * 2);
  unsigned short* gru1Rt  = (unsigned short*)alloc((size_t)1536 * 512 * 2);
  unsigned short* gru2Kt  = (unsigned short*)alloc((size_t)1536 * 512 * 2);
  unsigned short* gru2Rt  = (unsigned short*)alloc((size_t)1536 * 512 * 2);
  unsigned short* projWt  = (unsigned short*)alloc((size_t)80 * 512 * 2);
  unsigned short* h1_bf   = (unsigned short*)alloc((size_t)BT * 512 * 2);
  unsigned short* xp_bf   = (unsigned short*)alloc((size_t)BT * 256 * 2);
  float*          keys_f  = (float*)alloc((size_t)BM * 512 * 4);
  float*          x_f     = (float*)alloc((size_t)BT * 512 * 4);
  unsigned short* x_bf    = (unsigned short*)alloc((size_t)BT * 512 * 2);
  float*          s1_f    = (float*)alloc((size_t)BT * 512 * 4);
  unsigned short* s1_bf   = (unsigned short*)alloc((size_t)BT * 512 * 2);
  unsigned short* s2_bf   = (unsigned short*)alloc((size_t)BT * 512 * 2);
  float*          xz      = (float*)alloc((size_t)BT * 1536 * 4);
  float*          scr     = (float*)alloc((size_t)32 * 2048 * 4);

  auto nblk = [](long n) { return dim3((unsigned)((n + 255) / 256)); };

  // ---- weight / input prep (bf16, transposed)
  k_cvt_pad<<<nblk((long)BT * 96), 256, 0, stream>>>(inputs, inp_pad, BT, 80, 96);
  k_cvt_pad<<<nblk((long)BM * 256), 256, 0, stream>>>(attended, att_bf, BM, 256, 256);
  k_transpose_w<<<nblk(512L * 96), 256, 0, stream>>>(pre_W1, preW1t, 80, 512, 96, 0, 96);
  k_transpose_w<<<nblk(256L * 512), 256, 0, stream>>>(pre_W2, preW2t, 512, 256, 512, 0, 512);
  k_transpose_w<<<nblk(512L * 256), 256, 0, stream>>>(key_W, keyWt, 256, 512, 256, 0, 256);
  k_transpose_w<<<nblk(512L * 512), 256, 0, stream>>>(query_W, queryWt, 512, 512, 512, 0, 512);
  k_transpose_w<<<nblk(2048L * 512), 256, 0, stream>>>(lstm_K, lstmWt, 512, 2048, 1024, 0, 512);
  k_transpose_w<<<nblk(2048L * 512), 256, 0, stream>>>(lstm_R, lstmWt, 512, 2048, 1024, 512, 512);
  k_transpose_w<<<nblk(1536L * 512), 256, 0, stream>>>(gru1_K, gru1Kt, 512, 1536, 512, 0, 512);
  k_transpose_w<<<nblk(1536L * 512), 256, 0, stream>>>(gru1_R, gru1Rt, 512, 1536, 512, 0, 512);
  k_transpose_w<<<nblk(1536L * 512), 256, 0, stream>>>(gru2_K, gru2Kt, 512, 1536, 512, 0, 512);
  k_transpose_w<<<nblk(1536L * 512), 256, 0, stream>>>(gru2_R, gru2Rt, 512, 1536, 512, 0, 512);
  k_transpose_w<<<nblk(80L * 512), 256, 0, stream>>>(proj_W, projWt, 512, 80, 512, 0, 512);

  // ---- parallel GEMMs (prenet, keys)
  k_gemm_bf16<<<dim3(8, 100), 256, 0, stream>>>(inp_pad, 96, preW1t, 96, pre_b1,
                                                nullptr, h1_bf, BT, 512, 96, 1);
  k_gemm_bf16<<<dim3(4, 100), 256, 0, stream>>>(h1_bf, 512, preW2t, 512, pre_b2,
                                                nullptr, xp_bf, BT, 256, 512, 1);
  k_gemm_bf16<<<dim3(8, 50), 256, 0, stream>>>(att_bf, 256, keyWt, 256, key_b,
                                               keys_f, nullptr, BM, 512, 256, 0);

  // ---- attention + LSTM scan (single WGP, 32 waves)
  size_t smem1 = (size_t)32 * 1024 * 2 + (size_t)32 * 512 * 4 + (size_t)32 * 200 * 4;
  k_lstm_attn_scan<<<dim3(1), dim3(1024), smem1, stream>>>(
      xp_bf, keys_f, attended, queryWt, query_b, attnv_W, attnv_b,
      lstmWt, lstm_b, scr, x_f, x_bf);

  // ---- GRU1
  k_gemm_bf16<<<dim3(24, 100), 256, 0, stream>>>(x_bf, 512, gru1Kt, 512, gru1_b,
                                                 xz, nullptr, BT, 1536, 512, 0);
  size_t smem2 = (size_t)32 * 512 * 2;
  k_gru_scan<<<dim3(1), dim3(1024), smem2, stream>>>(xz, gru1Rt, gru1_b + 1536,
                                                     x_f, scr, s1_f, s1_bf);
  // ---- GRU2
  k_gemm_bf16<<<dim3(24, 100), 256, 0, stream>>>(s1_bf, 512, gru2Kt, 512, gru2_b,
                                                 xz, nullptr, BT, 1536, 512, 0);
  k_gru_scan<<<dim3(1), dim3(1024), smem2, stream>>>(xz, gru2Rt, gru2_b + 1536,
                                                     s1_f, scr, nullptr, s2_bf);

  // ---- final projection: mel = relu((x+y1+y2) @ proj_W)
  k_gemm_bf16<<<dim3(2, 100), 256, 0, stream>>>(s2_bf, 512, projWt, 512, nullptr,
                                                (float*)d_out, nullptr, BT, 80, 512, 1);
}
